// SkipGram_65515431133329
// MI455X (gfx1250) — matrix operand order repaired and checked by reading, not compiled, not run on hardware
//
#include <hip/hip_runtime.h>
#include <hip/hip_bf16.h>

#define EMB    512
#define BATCH  16384
#define NNEG   20
#define TILE   16
#define NTILES (BATCH / TILE)   // 1024
#define NWAVES 8                // waves per block
#define NPASS  (1 + NNEG)       // pass 0 = positive, 1..20 = negatives

typedef __attribute__((ext_vector_type(16))) __bf16 v16bf;
typedef __attribute__((ext_vector_type(8)))  float  v8f;

union Frag {
    v16bf v;
    unsigned int u[8];
    uint4 q[2];
};
union Acc {
    v8f   v;
    float f[8];
};

// f32 -> bf16 (round-half-up) and pack two into one dword: 3 VALU ops total
// result = (bf(lo)) | (bf(hi) << 16) via v_perm_b32
__device__ __forceinline__ unsigned int pack2bf(float lo, float hi) {
    union { float f; unsigned int u; } a, b;
    a.f = lo; b.f = hi;
    return __builtin_amdgcn_perm(b.u + 0x8000u, a.u + 0x8000u, 0x07060302u);
}

// B-matrix fragment for K-chunk c (K = 32c..32c+31); column = this lane's v-row.
// Lanes 0-15 hold K 32c+0..7,16..23; lanes 16-31 hold K 32c+8..15,24..31 (sequential pairs).
__device__ __forceinline__ v16bf load_bfrag(const float* __restrict__ vrow, int c, int hiHalf) {
    const float* p = vrow + 32 * c + (hiHalf ? 8 : 0);
    float4 x0 = *(const float4*)(p + 0);
    float4 x1 = *(const float4*)(p + 4);
    float4 x2 = *(const float4*)(p + 16);
    float4 x3 = *(const float4*)(p + 20);
    Frag b;
    b.u[0] = pack2bf(x0.x, x0.y); b.u[1] = pack2bf(x0.z, x0.w);
    b.u[2] = pack2bf(x1.x, x1.y); b.u[3] = pack2bf(x1.z, x1.w);
    b.u[4] = pack2bf(x2.x, x2.y); b.u[5] = pack2bf(x2.z, x2.w);
    b.u[6] = pack2bf(x3.x, x3.y); b.u[7] = pack2bf(x3.z, x3.w);
    return b.v;
}

__global__ void __launch_bounds__(32 * NWAVES)
sg_tile_kernel(const float* __restrict__ uw, const float* __restrict__ vw,
               const int* __restrict__ pos_u, const int* __restrict__ pos_v,
               const int* __restrict__ neg_v, float* __restrict__ partial)
{
    // A matrix (16x512 bf16) stored in WMMA fragment layout:
    // chunk c (K=32c..32c+31) occupies dwords [c*256, c*256+256);
    // lane L's 8 fragment dwords live at c*256 + L*8 .. +8.
    __shared__ unsigned int smA[16 * 256];   // 16 KB
    __shared__ float smPart[NWAVES];

    const int tid  = threadIdx.x;            // 0..255
    const int tile = blockIdx.x;             // 0..NTILES-1

    // ---------- Phase 1: cooperative gather + convert of 16 u-rows ----------
    {
        const int r = tid >> 4;              // row within tile (0..15)
        const int c = tid & 15;              // K-chunk (0..15)
        const float* urow = uw + (size_t)pos_u[tile * TILE + r] * EMB;
        const float* p = urow + 32 * c;      // this thread's 128 B slice
        float4 x0 = *(const float4*)(p + 0);
        float4 x1 = *(const float4*)(p + 4);
        float4 x2 = *(const float4*)(p + 8);
        float4 x3 = *(const float4*)(p + 12);
        float4 x4 = *(const float4*)(p + 16);
        float4 x5 = *(const float4*)(p + 20);
        float4 x6 = *(const float4*)(p + 24);
        float4 x7 = *(const float4*)(p + 28);

        unsigned int* dLo = &smA[c * 256 + r * 8];          // lane r
        unsigned int* dHi = &smA[c * 256 + (r + 16) * 8];   // lane r+16
        uint4 w;
        // lane r: dw0-3 = K 0..7, dw4-7 = K 16..23
        w.x = pack2bf(x0.x, x0.y); w.y = pack2bf(x0.z, x0.w);
        w.z = pack2bf(x1.x, x1.y); w.w = pack2bf(x1.z, x1.w);
        *(uint4*)(dLo + 0) = w;
        w.x = pack2bf(x4.x, x4.y); w.y = pack2bf(x4.z, x4.w);
        w.z = pack2bf(x5.x, x5.y); w.w = pack2bf(x5.z, x5.w);
        *(uint4*)(dLo + 4) = w;
        // lane r+16: dw0-3 = K 8..15, dw4-7 = K 24..31
        w.x = pack2bf(x2.x, x2.y); w.y = pack2bf(x2.z, x2.w);
        w.z = pack2bf(x3.x, x3.y); w.w = pack2bf(x3.z, x3.w);
        *(uint4*)(dHi + 0) = w;
        w.x = pack2bf(x6.x, x6.y); w.y = pack2bf(x6.z, x6.w);
        w.z = pack2bf(x7.x, x7.y); w.w = pack2bf(x7.z, x7.w);
        *(uint4*)(dHi + 4) = w;
    }
    __syncthreads();

    // ---------- Phase 2: 8 waves split the 21 dot-product passes ----------
    const int wave   = tid >> 5;             // 0..7
    const int lane   = tid & 31;             // 0..31
    const int l      = lane & 15;
    const int hiHalf = (lane >> 4) & 1;
    const int bcol   = tile * TILE + l;      // batch element this lane-pair owns

    float loss = 0.0f;
    for (int pass = wave; pass < NPASS; pass += NWAVES) {
        const int idx = (pass == 0) ? pos_v[bcol]
                                    : neg_v[(size_t)bcol * NNEG + (pass - 1)];
        const float* vrow = vw + (size_t)idx * EMB;

        Acc acc;
#pragma unroll
        for (int i = 0; i < 8; ++i) acc.f[i] = 0.0f;
#pragma unroll
        for (int c = 0; c < 16; ++c) {
            Frag a;
            const uint4* ap = (const uint4*)&smA[c * 256 + lane * 8];
            a.q[0] = ap[0];
            a.q[1] = ap[1];
            v16bf bf = load_bfrag(vrow, c, hiHalf);
            acc.v = __builtin_amdgcn_wmma_f32_16x16x32_bf16(
                /*neg_a=*/false, a.v, /*neg_b=*/false, bf,
                /*c_mod=*/(short)0, acc.v, /*reuse_a=*/false, /*reuse_b=*/false);
        }
        // diagonal (d,d): lanes 0-7 -> acc.f[lane], lanes 24-31 -> acc.f[lane-24]
        const int r = (lane < 8) ? lane : (lane - 24);
        float s = acc.f[0];
#pragma unroll
        for (int i = 1; i < 8; ++i)
            if (r == i) s = acc.f[i];
        s = fminf(10.0f, fmaxf(-10.0f, s));
        // pass 0: -logsig(s) = log1p(exp(-s)); negatives: -logsig(-s) = log1p(exp(s))
        loss += log1pf(expf((pass == 0) ? -s : s));
    }

    // only diagonal-owning lanes carry a real per-batch loss
    if (!((lane < 8) || (lane >= 24))) loss = 0.0f;

    // wave32 butterfly reduction, then block reduction of the 8 wave partials
#pragma unroll
    for (int off = 16; off >= 1; off >>= 1)
        loss += __shfl_xor(loss, off, 32);
    if (lane == 0) smPart[wave] = loss;
    __syncthreads();
    if (tid == 0) {
        float s = 0.0f;
#pragma unroll
        for (int w = 0; w < NWAVES; ++w) s += smPart[w];
        partial[tile] = s;
    }
}

__global__ void __launch_bounds__(256)
sg_reduce_kernel(const float* __restrict__ partial, float* __restrict__ out, int n)
{
    __shared__ float sm[256];
    float s = 0.0f;
    for (int i = threadIdx.x; i < n; i += 256) s += partial[i];
    sm[threadIdx.x] = s;
    __syncthreads();
#pragma unroll
    for (int st = 128; st >= 1; st >>= 1) {
        if (threadIdx.x < st) sm[threadIdx.x] += sm[threadIdx.x + st];
        __syncthreads();
    }
    if (threadIdx.x == 0) out[0] = sm[0] * (1.0f / (float)BATCH);
}

extern "C" void kernel_launch(void* const* d_in, const int* in_sizes, int n_in,
                              void* d_out, int out_size, void* d_ws, size_t ws_size,
                              hipStream_t stream)
{
    (void)in_sizes; (void)n_in; (void)out_size; (void)ws_size;
    const float* u_weight = (const float*)d_in[0];
    const float* v_weight = (const float*)d_in[1];
    const int*   pos_u    = (const int*)d_in[2];
    const int*   pos_v    = (const int*)d_in[3];
    const int*   neg_v    = (const int*)d_in[4];
    float*       out      = (float*)d_out;
    float*       partial  = (float*)d_ws;   // NTILES floats of scratch

    sg_tile_kernel<<<NTILES, 32 * NWAVES, 0, stream>>>(u_weight, v_weight,
                                                       pos_u, pos_v, neg_v, partial);
    sg_reduce_kernel<<<1, 256, 0, stream>>>(partial, out, NTILES);
}
